// LSHAttention_43164421325472
// MI455X (gfx1250) — matrix operand order repaired and checked by reading, not compiled, not run on hardware
//
#include <hip/hip_runtime.h>

// ---------------------------------------------------------------------------
// LSH attention for MI455X (gfx1250), bf16 WMMA throughout.
// Pipeline: cvt -> proj GEMMs (WMMA) -> hash -> fused flash-LSH attention
// (WMMA QK^T + online masked softmax + WMMA P*V) -> output GEMM (WMMA).
// All intermediates ~6 MB => L2 resident; dots never materialized.
// This round: explicit cross-iteration software pipelining (double-buffered
// fragments) so s_wait_loadcnt waits land a full iteration after issue.
// ---------------------------------------------------------------------------

typedef __bf16 bf16;
typedef __attribute__((ext_vector_type(8)))  bf16  v8bf;
typedef __attribute__((ext_vector_type(16))) bf16  v16bf;
typedef __attribute__((ext_vector_type(8)))  float v8f;

#define S_TOK 1024
#define DMDL  512
#define NHEAD 8
#define HDIM  64
#define NROWS 8192   // B*H*S
#define NHASH 8
#define NROT  8

static __device__ __forceinline__ v16bf ldfrag(const bf16* p0, const bf16* p1) {
  v8bf lo = *(const v8bf*)p0;
  v8bf hi = *(const v8bf*)p1;
  return __builtin_shufflevector(lo, hi, 0,1,2,3,4,5,6,7,8,9,10,11,12,13,14,15);
}

static __device__ __forceinline__ v8f wmma_bf16(v16bf a, v16bf b, v8f c) {
  // D = A(16x32 bf16) * B(32x16 bf16) + C(16x16 f32)
  return __builtin_amdgcn_wmma_f32_16x16x32_bf16(false, a, false, b, (short)0, c,
                                                 false, false);
}

// Load the 4 K-score B-fragments (32 cols x K=64) for column tile j.
static __device__ __forceinline__ void load_k4(const bf16* __restrict__ QK,
                                               int j, int ln, int bkb,
                                               v16bf kf[4]) {
  const bf16* b0 = QK + (size_t)(j + ln) * HDIM + bkb;
  const bf16* b1 = QK + (size_t)(j + 16 + ln) * HDIM + bkb;
  kf[0] = ldfrag(b0,      b0 + 8);
  kf[1] = ldfrag(b0 + 32, b0 + 40);
  kf[2] = ldfrag(b1,      b1 + 8);
  kf[3] = ldfrag(b1 + 32, b1 + 40);
}

// --------------------------- conversion kernels ----------------------------

__global__ void k_cvt(const float* __restrict__ src, bf16* __restrict__ dst, int n) {
  int i = blockIdx.x * blockDim.x + threadIdx.x;
  if (i < n) dst[i] = (bf16)src[i];
}

// w (K x N, row-major f32) -> wT (N x K, row-major bf16): K contiguous per col
__global__ void k_cvt_t(const float* __restrict__ w, bf16* __restrict__ wT) {
  int i = blockIdx.x * blockDim.x + threadIdx.x;
  if (i < DMDL * DMDL) {
    int k = i / DMDL, n = i % DMDL;
    wT[n * DMDL + k] = (bf16)w[i];
  }
}

// --------------------------- projection GEMMs ------------------------------
// out(1024x512) = A(1024x512 bf16) @ BT^T + bias. One wave => 16x64 tile.
// mode 0: store qk layout   qk[(c>>6)*1024 + m][c&63]      (bf16, N x 64)
// mode 1: store vT layout   vT[c&63][(c>>6)*1024 + m]      (bf16, 64 x N)
// mode 2: store f32 row-major out[m*512 + c]               (final output)

__global__ void __launch_bounds__(256) k_gemm(const bf16* __restrict__ A,
                                              const bf16* __restrict__ BT,
                                              const float* __restrict__ bias,
                                              void* __restrict__ out, int mode) {
  const int lane = threadIdx.x & 31;
  const int wid  = (blockIdx.x * blockDim.x + threadIdx.x) >> 5;   // 0..511
  const int m0 = (wid >> 3) * 16;
  const int c0 = (wid & 7) * 64;
  const int ln = lane & 15, hs = lane >> 4;
  const int akb = hs * 16, bkb = hs * 16;

  v8f acc[4] = {};
  const bf16* arow = A + (size_t)(m0 + ln) * DMDL;
  const bf16* brow[4];
  #pragma unroll
  for (int i = 0; i < 4; ++i) brow[i] = BT + (size_t)(c0 + i * 16 + ln) * DMDL + bkb;

  // software pipeline: double-buffered fragments
  v16bf a = ldfrag(arow + akb, arow + akb + 8);
  v16bf bfr[4];
  #pragma unroll
  for (int i = 0; i < 4; ++i) bfr[i] = ldfrag(brow[i], brow[i] + 8);

  #pragma unroll 2
  for (int k0 = 0; k0 < DMDL; k0 += 32) {
    const int kn = (k0 + 32) & (DMDL - 1);   // wraps on last trip (harmless)
    v16bf an = ldfrag(arow + kn + akb, arow + kn + akb + 8);
    v16bf bn[4];
    #pragma unroll
    for (int i = 0; i < 4; ++i) bn[i] = ldfrag(brow[i] + kn, brow[i] + kn + 8);
    #pragma unroll
    for (int i = 0; i < 4; ++i) acc[i] = wmma_bf16(a, bfr[i], acc[i]);
    a = an;
    #pragma unroll
    for (int i = 0; i < 4; ++i) bfr[i] = bn[i];
  }

  #pragma unroll
  for (int i = 0; i < 4; ++i) {
    #pragma unroll
    for (int g = 0; g < 8; ++g) {
      int m = m0 + g + hs * 8;
      int c = c0 + i * 16 + ln;
      float val = acc[i][g] + bias[c];
      if (mode == 0) {
        int n = (c >> 6) * S_TOK + m;                 // head-major row index
        ((bf16*)out)[(size_t)n * HDIM + (c & 63)] = (bf16)val;
      } else if (mode == 1) {
        ((bf16*)out)[(size_t)(c & 63) * NROWS + (c >> 6) * S_TOK + m] = (bf16)val;
      } else {
        ((float*)out)[(size_t)m * DMDL + c] = val;
      }
    }
  }
}

// ------------------------------- LSH hashing -------------------------------

__global__ void k_hash(const bf16* __restrict__ QK, const float* __restrict__ rot,
                       int* __restrict__ BK) {
  int idx = blockIdx.x * blockDim.x + threadIdx.x;   // 65536 = 8 rounds * 8192
  int r = idx >> 13, n = idx & (NROWS - 1);
  int h = n >> 10;
  float rv[NROT] = {};
  const float* rb = rot + (size_t)((r * NHEAD + h) * HDIM) * NROT;
  #pragma unroll 8
  for (int d = 0; d < HDIM; ++d) {
    float q = (float)QK[(size_t)n * HDIM + d];
    #pragma unroll
    for (int c = 0; c < NROT; ++c) rv[c] = fmaf(q, rb[d * NROT + c], rv[c]);
  }
  float best = rv[0]; int bi = 0;
  #pragma unroll
  for (int c = 1; c < NROT; ++c) if (rv[c] > best) { best = rv[c]; bi = c; }
  #pragma unroll
  for (int c = 0; c < NROT; ++c) if (-rv[c] > best) { best = -rv[c]; bi = NROT + c; }
  BK[r * NROWS + n] = bi;
}

// -------------------- fused flash-style LSH attention ----------------------
// One wave owns 16 query rows. For each of 8 hash rounds: stream all 8192
// keys/values in 32-col tiles; scores via 4 bf16 WMMAs; bucket mask at -1e30;
// online softmax (shfl_xor reductions within 16-lane halves); P goes through
// a 1KB per-wave LDS staging buffer to re-layout C->A; P*V via 4 bf16 WMMAs.
// K-score fragments are double-buffered across iterations: next-tile loads
// issue right after the current score WMMAs and complete during the softmax
// VALU block + LDS round-trip + P*V WMMAs.

__global__ void __launch_bounds__(256) k_attn(const bf16* __restrict__ QK,
                                              const bf16* __restrict__ VT,
                                              const int* __restrict__ BK,
                                              bf16* __restrict__ ctx) {
  __shared__ bf16 plds[8][16 * 32];
  const int lane = threadIdx.x & 31;
  const int wv   = threadIdx.x >> 5;
  const int row0 = (blockIdx.x * 8 + wv) * 16;
  const int ln = lane & 15, hs = lane >> 4;
  const int akb = hs * 16, bkb = hs * 16;

  const bf16* qrow = QK + (size_t)(row0 + ln) * HDIM;
  v16bf qa0 = ldfrag(qrow + akb,      qrow + akb + 8);
  v16bf qa1 = ldfrag(qrow + 32 + akb, qrow + 32 + akb + 8);

  v8f tot[4] = {};
  #pragma unroll 1
  for (int r = 0; r < NHASH; ++r) {
    const int* bkr = BK + r * NROWS;
    int rowb[8]; float m_[8], l_[8];
    #pragma unroll
    for (int g = 0; g < 8; ++g) {
      rowb[g] = bkr[row0 + g + hs * 8];
      m_[g] = -1e30f; l_[g] = 0.f;
    }
    v8f acc[4] = {};

    // pipeline prologue: K fragments + buckets for tile j=0
    v16bf kf[4];
    load_k4(QK, 0, ln, bkb, kf);
    int cb0 = bkr[ln], cb1 = bkr[16 + ln];

    #pragma unroll 1
    for (int j = 0; j < NROWS; j += 32) {
      const int jn = (j + 32) & (NROWS - 1);   // wraps on last trip (harmless)

      // V fragments for the current tile (consumed at the bottom)
      v16bf vf[4];
      #pragma unroll
      for (int t4 = 0; t4 < 4; ++t4) {
        const bf16* vb = VT + (size_t)(t4 * 16 + ln) * NROWS + j + bkb;
        vf[t4] = ldfrag(vb, vb + 8);
      }

      // ---- scores (current kf, loaded one iteration ago) ----
      v8f s0 = {}, s1 = {};
      s0 = wmma_bf16(qa0, kf[0], s0);
      s0 = wmma_bf16(qa1, kf[1], s0);
      s1 = wmma_bf16(qa0, kf[2], s1);
      s1 = wmma_bf16(qa1, kf[3], s1);

      // ---- issue next-tile K fragment + bucket loads now ----
      v16bf kn[4];
      load_k4(QK, jn, ln, bkb, kn);
      int cbn0 = bkr[jn + ln], cbn1 = bkr[jn + 16 + ln];

      // ---- masked online softmax (acc rescale fused per row) ----
      #pragma unroll
      for (int g = 0; g < 8; ++g) {
        float a0 = (rowb[g] == cb0) ? s0[g] * 0.125f : -1e30f;  // 1/sqrt(64)
        float a1 = (rowb[g] == cb1) ? s1[g] * 0.125f : -1e30f;
        float t = fmaxf(a0, a1);
        t = fmaxf(t, __shfl_xor(t, 1));
        t = fmaxf(t, __shfl_xor(t, 2));
        t = fmaxf(t, __shfl_xor(t, 4));
        t = fmaxf(t, __shfl_xor(t, 8));
        float mn = fmaxf(m_[g], t);
        float alpha = __expf(m_[g] - mn);
        m_[g] = mn;
        a0 = __expf(a0 - mn);
        a1 = __expf(a1 - mn);
        s0[g] = a0; s1[g] = a1;
        float sum = a0 + a1;
        sum += __shfl_xor(sum, 1);
        sum += __shfl_xor(sum, 2);
        sum += __shfl_xor(sum, 4);
        sum += __shfl_xor(sum, 8);
        l_[g] = l_[g] * alpha + sum;
        #pragma unroll
        for (int t4 = 0; t4 < 4; ++t4) acc[t4][g] *= alpha;
      }

      // ---- P (C-layout f32) -> LDS bf16 16x32, reload as A-frag ----
      bf16* wl = &plds[wv][0];
      #pragma unroll
      for (int g = 0; g < 8; ++g) {
        wl[(g + hs * 8) * 32 + ln]      = (bf16)s0[g];
        wl[(g + hs * 8) * 32 + 16 + ln] = (bf16)s1[g];
      }
      asm volatile("s_wait_dscnt 0x0" ::: "memory");  // per-wave LDS RAW fence
      const bf16* pr = wl + ln * 32 + akb;
      v16bf pa = ldfrag(pr, pr + 8);

      // ---- P @ V ----
      #pragma unroll
      for (int t4 = 0; t4 < 4; ++t4) acc[t4] = wmma_bf16(pa, vf[t4], acc[t4]);

      // ---- rotate pipeline buffers ----
      #pragma unroll
      for (int i = 0; i < 4; ++i) kf[i] = kn[i];
      cb0 = cbn0; cb1 = cbn1;
    }
    #pragma unroll
    for (int g = 0; g < 8; ++g) l_[g] = 1.f / l_[g];
    #pragma unroll
    for (int t4 = 0; t4 < 4; ++t4)
      #pragma unroll
      for (int g = 0; g < 8; ++g) tot[t4][g] += acc[t4][g] * l_[g];
  }
  // ctx stored already in (s, h*64+hd) order for the output GEMM, / NHASH
  #pragma unroll
  for (int t4 = 0; t4 < 4; ++t4)
    #pragma unroll
    for (int g = 0; g < 8; ++g) {
      int n = row0 + g + hs * 8;
      int h = n >> 10, s = n & (S_TOK - 1);
      ctx[(size_t)s * DMDL + h * HDIM + t4 * 16 + ln] =
          (bf16)(tot[t4][g] * (1.f / NHASH));
    }
}

// ------------------------------- launcher ----------------------------------

extern "C" void kernel_launch(void* const* d_in, const int* in_sizes, int n_in,
                              void* d_out, int out_size, void* d_ws, size_t ws_size,
                              hipStream_t stream) {
  (void)in_sizes; (void)n_in; (void)out_size; (void)ws_size;
  const float* x    = (const float*)d_in[0];
  const float* w_qk = (const float*)d_in[1];
  const float* b_qk = (const float*)d_in[2];
  const float* w_v  = (const float*)d_in[3];
  const float* b_v  = (const float*)d_in[4];
  const float* w_o  = (const float*)d_in[5];
  const float* b_o  = (const float*)d_in[6];
  const float* rot  = (const float*)d_in[7];

  char* ws = (char*)d_ws;
  size_t off = 0;
  bf16* xbf  = (bf16*)(ws + off); off += (size_t)S_TOK * DMDL * 2;   // 1 MB
  bf16* wqkT = (bf16*)(ws + off); off += (size_t)DMDL * DMDL * 2;    // 512 KB
  bf16* wvT  = (bf16*)(ws + off); off += (size_t)DMDL * DMDL * 2;
  bf16* woT  = (bf16*)(ws + off); off += (size_t)DMDL * DMDL * 2;
  bf16* qkb  = (bf16*)(ws + off); off += (size_t)NROWS * HDIM * 2;   // 1 MB
  bf16* vT   = (bf16*)(ws + off); off += (size_t)NROWS * HDIM * 2;   // 1 MB
  int*  bk   = (int*) (ws + off); off += (size_t)NHASH * NROWS * 4;  // 256 KB
  bf16* ctx  = (bf16*)(ws + off); off += (size_t)S_TOK * DMDL * 2;   // 1 MB

  k_cvt  <<<(S_TOK * DMDL) / 256, 256, 0, stream>>>(x, xbf, S_TOK * DMDL);
  k_cvt_t<<<(DMDL * DMDL) / 256, 256, 0, stream>>>(w_qk, wqkT);
  k_cvt_t<<<(DMDL * DMDL) / 256, 256, 0, stream>>>(w_v,  wvT);
  k_cvt_t<<<(DMDL * DMDL) / 256, 256, 0, stream>>>(w_o,  woT);

  k_gemm<<<64, 256, 0, stream>>>(xbf, wqkT, b_qk, (void*)qkb, 0);   // qk
  k_gemm<<<64, 256, 0, stream>>>(xbf, wvT,  b_v,  (void*)vT,  1);   // v^T

  k_hash<<<(NHASH * NROWS) / 256, 256, 0, stream>>>(qkb, rot, bk);

  k_attn<<<64, 256, 0, stream>>>(qkb, vT, bk, ctx);

  k_gemm<<<64, 256, 0, stream>>>(ctx, woT, b_o, d_out, 2);          // out proj
}
